// AttentionConv2d_44899588112400
// MI455X (gfx1250) — matrix operand order repaired and checked by reading, not compiled, not run on hardware
//
#include <hip/hip_runtime.h>

typedef __attribute__((ext_vector_type(16))) __bf16          v16bf;
typedef __attribute__((ext_vector_type(16))) unsigned short  v16u;
typedef __attribute__((ext_vector_type(8)))  float           v8f;

// ---------- bf16 helpers ----------
__device__ __forceinline__ unsigned short f2bf(float f) {
  unsigned int u = __builtin_bit_cast(unsigned int, f);
  u += 0x7FFFu + ((u >> 16) & 1u);          // round-to-nearest-even
  return (unsigned short)(u >> 16);
}

// A fragment: 16x32 bf16, row m = lane&15, half = lane>>4
// element e -> k = e + 16*half   (contiguous order, matches B fragment)
__device__ __forceinline__ v16bf load_a_frag(const unsigned short* base, int ld,
                                             int m, int k0, int hh) {
  v16u t;
#pragma unroll
  for (int e = 0; e < 16; ++e) {
    int k = e + 16 * hh;
    t[e] = base[m * ld + k0 + k];
  }
  return __builtin_bit_cast(v16bf, t);
}

// B fragment from B^T stored N x K row-major: col n = lane&15,
// element e -> k = e + 16*half  (contiguous order, same as the A fragment)
__device__ __forceinline__ v16bf load_bt_frag(const unsigned short* bt, int ld,
                                              int n, int k0, int hh) {
  v16u t;
#pragma unroll
  for (int e = 0; e < 16; ++e) t[e] = bt[n * ld + k0 + 16 * hh + e];
  return __builtin_bit_cast(v16bf, t);
}

__device__ __forceinline__ v8f wmma_bf16(v16bf a, v16bf b, v8f c) {
  return __builtin_amdgcn_wmma_f32_16x16x32_bf16(false, a, false, b, (short)0, c,
                                                 false, false);
}

// ---------- prep / conversion kernels ----------
__global__ void k_cvt(const float* __restrict__ src, unsigned short* __restrict__ dst, int n) {
  int i = blockIdx.x * blockDim.x + threadIdx.x;
  if (i < n) dst[i] = f2bf(src[i]);
}

// x (b,c,32,32) f32  ->  xT [b*1024 + l][c] bf16
__global__ void k_cvt_xT(const float* __restrict__ x, unsigned short* __restrict__ xT) {
  int i = blockIdx.x * blockDim.x + threadIdx.x;   // 1,048,576
  int c = i & 255, l = (i >> 8) & 1023, b = i >> 18;
  xT[i] = f2bf(x[(b * 256 + c) * 1024 + l]);
}

// key_rel (63,32) f32 -> bf16 padded to 64 rows (row 63 = 0)
__global__ void k_cvt_rel(const float* __restrict__ rel, unsigned short* __restrict__ dst) {
  int i = blockIdx.x * blockDim.x + threadIdx.x;   // 2048
  int r = i >> 5, d = i & 31;
  dst[i] = (r < 63) ? f2bf(rel[r * 32 + d]) : (unsigned short)0;
}

// ---------- qkv projection: [4096,768] = xT[4096,256] * wqkv^T, scatter q/k/v ----------
// fully-unrolled ping-pong pipeline: no loop-carried register copies
__global__ __launch_bounds__(128) void k_qkv(const unsigned short* __restrict__ xT,
                                             const unsigned short* __restrict__ wq,
                                             const float* __restrict__ bqkv,
                                             unsigned short* __restrict__ qbuf,
                                             unsigned short* __restrict__ kbuf,
                                             unsigned short* __restrict__ vbuf) {
  const int lane = threadIdx.x & 31, wv = threadIdx.x >> 5;
  const int hh = lane >> 4, ln15 = lane & 15;
  const int M0 = blockIdx.y * 64 + wv * 16;
  const int N0 = blockIdx.x * 64;
  const unsigned short* Arow = xT + M0 * 256;
  v8f acc[4] = {};
  v16bf a0 = load_a_frag(Arow, 256, ln15, 0, hh);
  v16bf b0[4], a1, b1[4];
#pragma unroll
  for (int nt = 0; nt < 4; ++nt)
    b0[nt] = load_bt_frag(wq + (N0 + nt * 16) * 256, 256, ln15, 0, hh);
#pragma unroll
  for (int k0 = 0; k0 < 256; k0 += 64) {
    a1 = load_a_frag(Arow, 256, ln15, k0 + 32, hh);
#pragma unroll
    for (int nt = 0; nt < 4; ++nt)
      b1[nt] = load_bt_frag(wq + (N0 + nt * 16) * 256, 256, ln15, k0 + 32, hh);
#pragma unroll
    for (int nt = 0; nt < 4; ++nt) acc[nt] = wmma_bf16(a0, b0[nt], acc[nt]);
    if (k0 + 64 < 256) {                       // statically resolved after unroll
      a0 = load_a_frag(Arow, 256, ln15, k0 + 64, hh);
#pragma unroll
      for (int nt = 0; nt < 4; ++nt)
        b0[nt] = load_bt_frag(wq + (N0 + nt * 16) * 256, 256, ln15, k0 + 64, hh);
    }
#pragma unroll
    for (int nt = 0; nt < 4; ++nt) acc[nt] = wmma_bf16(a1, b1[nt], acc[nt]);
  }
  const float scale = 0.17677669529663687f;   // 1/sqrt(DKH=32)
#pragma unroll
  for (int nt = 0; nt < 4; ++nt) {
    int o = N0 + nt * 16 + ln15;
    float bias = bqkv[o];
    int hq = (o >> 5) & 7, dd = o & 31;
#pragma unroll
    for (int r = 0; r < 8; ++r) {
      int gl = M0 + r + 8 * hh;            // global row = b*1024 + l
      int b = gl >> 10, l = gl & 1023;
      int bh = b * 8 + hq;
      float v = acc[nt][r] + bias;
      if (o < 256)       qbuf[(bh * 1024 + l) * 32 + dd] = f2bf(v * scale);
      else if (o < 512)  kbuf[(bh * 1024 + l) * 32 + dd] = f2bf(v);
      else               vbuf[(bh * 32 + dd) * 1024 + l] = f2bf(v);   // V^T [d][l]
    }
  }
}

// ---------- fused flash attention with relative logits ----------
// one wave per (b, h, 16-query tile); online softmax over 1024 keys, 32 at a time,
// 2x-unrolled ping-pong on the K fragments (no loop-carried copies)
__global__ __launch_bounds__(32) void k_attn(const unsigned short* __restrict__ qbuf,
                                             const unsigned short* __restrict__ kbuf,
                                             const unsigned short* __restrict__ vbuf,
                                             const unsigned short* __restrict__ relh,
                                             const unsigned short* __restrict__ relw,
                                             unsigned short* __restrict__ attnT) {
  __shared__ float phs[16 * 64];
  __shared__ float pws[16 * 64];
  __shared__ unsigned short pbuf[16 * 32];
  const int lane = threadIdx.x;
  const int hh = lane >> 4, ln15 = lane & 15;
  const int h = blockIdx.y, b = blockIdx.z;
  const int bh = b * 8 + h;
  const int l0 = blockIdx.x * 16;
  const int xr = l0 >> 5;        // query row, shared by all 16 queries of the tile
  const int y0 = l0 & 31;        // query col base; y(qi) = y0 + qi
  const unsigned short* krows = kbuf + bh * 1024 * 32;
  const unsigned short* vrow0 = vbuf + (bh * 32 + ln15) * 1024;        // d = ln15
  const unsigned short* vrow1 = vbuf + (bh * 32 + 16 + ln15) * 1024;   // d = 16+ln15

  // Q fragment (16 queries x 32 dims), reused by every WMMA below
  v16bf aq = load_a_frag(qbuf + (bh * 1024 + l0) * 32, 32, ln15, 0, hh);

  // rel-logit products P[l,r] = q_l . rel[r]  (16x64 each), via 4+4 WMMAs -> LDS
#pragma unroll
  for (int t = 0; t < 4; ++t) {
    v16bf bmh = load_bt_frag(relh + (t * 16) * 32, 32, ln15, 0, hh);
    v8f ch = {}; ch = wmma_bf16(aq, bmh, ch);
    v16bf bmw = load_bt_frag(relw + (t * 16) * 32, 32, ln15, 0, hh);
    v8f cw = {}; cw = wmma_bf16(aq, bmw, cw);
#pragma unroll
    for (int r = 0; r < 8; ++r) {
      phs[(r + 8 * hh) * 64 + t * 16 + ln15] = ch[r];
      pws[(r + 8 * hh) * 64 + t * 16 + ln15] = cw[r];
    }
  }
  __syncthreads();

  v8f acc0 = {}, acc1 = {};
  float mrun[8], srun[8];
#pragma unroll
  for (int r = 0; r < 8; ++r) { mrun[r] = -1e30f; srun[r] = 0.f; }

  // one 32-key chunk: logits WMMAs + rel add + online softmax + PV WMMAs
  auto step = [&](int mc, v16bf bk0, v16bf bk1) {
    v8f c0 = {}, c1 = {};
    c0 = wmma_bf16(aq, bk0, c0);
    c1 = wmma_bf16(aq, bk1, c1);
    const int m0k = mc + ln15, m1k = mc + 16 + ln15;
    const int i0 = m0k >> 5, j0 = m0k & 31;
    const int i1 = m1k >> 5, j1 = m1k & 31;
#pragma unroll
    for (int r = 0; r < 8; ++r) {
      int qi = r + 8 * hh;
      int yq = y0 + qi;
      float v0 = c0[r] + phs[qi * 64 + (i0 - xr + 31)] + pws[qi * 64 + (j0 - yq + 31)];
      float v1 = c1[r] + phs[qi * 64 + (i1 - xr + 31)] + pws[qi * 64 + (j1 - yq + 31)];
      // row max across the 16 lanes of this half-wave (C layout: row lives in one half)
      float mx = fmaxf(v0, v1);
#pragma unroll
      for (int off = 8; off >= 1; off >>= 1) mx = fmaxf(mx, __shfl_xor(mx, off, 32));
      float mnew = fmaxf(mrun[r], mx);
      float alpha = __expf(mrun[r] - mnew);
      mrun[r] = mnew;
      float p0 = __expf(v0 - mnew), p1 = __expf(v1 - mnew);
      float ps = p0 + p1;
#pragma unroll
      for (int off = 8; off >= 1; off >>= 1) ps += __shfl_xor(ps, off, 32);
      srun[r] = srun[r] * alpha + ps;
      acc0[r] *= alpha;
      acc1[r] *= alpha;
      pbuf[qi * 32 + ln15]      = f2bf(p0);   // LDS transpose C-layout -> A-layout
      pbuf[qi * 32 + 16 + ln15] = f2bf(p1);
    }
    __syncthreads();
    v16bf ap  = load_a_frag(pbuf, 32, ln15, 0, hh);
    v16bf bv0 = load_bt_frag(vbuf + bh * 32 * 1024, 1024, ln15, mc, hh);        // d 0..15
    v16bf bv1 = load_bt_frag(vbuf + (bh * 32 + 16) * 1024, 1024, ln15, mc, hh); // d 16..31
    acc0 = wmma_bf16(ap, bv0, acc0);
    acc1 = wmma_bf16(ap, bv1, acc1);
    __syncthreads();
  };

  // prime chunk 0, then ping-pong two chunks per iteration
  v16bf ka0 = load_bt_frag(krows, 32, ln15, 0, hh);
  v16bf kb0 = load_bt_frag(krows + 16 * 32, 32, ln15, 0, hh);
  for (int mc = 0; mc < 1024; mc += 64) {
    v16bf ka1 = load_bt_frag(krows + (mc + 32) * 32, 32, ln15, 0, hh);
    v16bf kb1 = load_bt_frag(krows + (mc + 48) * 32, 32, ln15, 0, hh);
    __builtin_prefetch(vrow0 + mc + 32, 0, 1);   // global_prefetch_b8
    __builtin_prefetch(vrow1 + mc + 32, 0, 1);
    step(mc, ka0, kb0);
    const int mn = (mc + 64) & 1023;             // wraps harmlessly on last iter
    ka0 = load_bt_frag(krows + mn * 32, 32, ln15, 0, hh);
    kb0 = load_bt_frag(krows + (mn + 16) * 32, 32, ln15, 0, hh);
    __builtin_prefetch(vrow0 + mn, 0, 1);
    __builtin_prefetch(vrow1 + mn, 0, 1);
    step(mc + 32, ka1, kb1);
  }

  // normalize, store head output as bf16 [b*1024+l][c = h*32+d]
#pragma unroll
  for (int r = 0; r < 8; ++r) {
    int qi = r + 8 * hh;
    float inv = 1.0f / srun[r];
    int base = (b * 1024 + l0 + qi) * 256 + h * 32;
    attnT[base + ln15]      = f2bf(acc0[r] * inv);
    attnT[base + 16 + ln15] = f2bf(acc1[r] * inv);
  }
}

// ---------- fused output: out[:, :256] = x*w_out^T + b_out; out[:, 256:] = attn*w_attn^T + b_attn
__global__ __launch_bounds__(128) void k_out(const unsigned short* __restrict__ xT,
                                             const unsigned short* __restrict__ attnT,
                                             const unsigned short* __restrict__ wout,
                                             const unsigned short* __restrict__ wattn,
                                             const float* __restrict__ bout,
                                             const float* __restrict__ battn,
                                             float* __restrict__ out) {
  const int lane = threadIdx.x & 31, wv = threadIdx.x >> 5;
  const int hh = lane >> 4, ln15 = lane & 15;
  const int M0 = blockIdx.y * 64 + wv * 16;
  const int N0 = blockIdx.x * 64;                 // 0..511
  const bool conv = (N0 < 256);
  const unsigned short* A  = conv ? xT : attnT;
  const unsigned short* Wb = conv ? wout : wattn;
  const float* bias        = conv ? bout : battn;
  const int nloc0          = conv ? N0 : (N0 - 256);
  const unsigned short* Arow = A + M0 * 256;
  v8f acc[4] = {};
  v16bf a0 = load_a_frag(Arow, 256, ln15, 0, hh);
  v16bf b0[4], a1, b1[4];
#pragma unroll
  for (int nt = 0; nt < 4; ++nt)
    b0[nt] = load_bt_frag(Wb + (nloc0 + nt * 16) * 256, 256, ln15, 0, hh);
#pragma unroll
  for (int k0 = 0; k0 < 256; k0 += 64) {
    a1 = load_a_frag(Arow, 256, ln15, k0 + 32, hh);
#pragma unroll
    for (int nt = 0; nt < 4; ++nt)
      b1[nt] = load_bt_frag(Wb + (nloc0 + nt * 16) * 256, 256, ln15, k0 + 32, hh);
#pragma unroll
    for (int nt = 0; nt < 4; ++nt) acc[nt] = wmma_bf16(a0, b0[nt], acc[nt]);
    if (k0 + 64 < 256) {                       // statically resolved after unroll
      a0 = load_a_frag(Arow, 256, ln15, k0 + 64, hh);
#pragma unroll
      for (int nt = 0; nt < 4; ++nt)
        b0[nt] = load_bt_frag(Wb + (nloc0 + nt * 16) * 256, 256, ln15, k0 + 64, hh);
    }
#pragma unroll
    for (int nt = 0; nt < 4; ++nt) acc[nt] = wmma_bf16(a1, b1[nt], acc[nt]);
  }
#pragma unroll
  for (int nt = 0; nt < 4; ++nt) {
    int o = N0 + nt * 16 + ln15;
    float bs = bias[conv ? o : (o - 256)];
#pragma unroll
    for (int r = 0; r < 8; ++r) {
      int gl = M0 + r + 8 * hh;
      int bb = gl >> 10, l = gl & 1023;
      out[(bb * 512 + o) * 1024 + l] = acc[nt][r] + bs;
    }
  }
}

// ---------- host ----------
extern "C" void kernel_launch(void* const* d_in, const int* in_sizes, int n_in,
                              void* d_out, int out_size, void* d_ws, size_t ws_size,
                              hipStream_t stream) {
  const float* x     = (const float*)d_in[0];
  const float* wqkv  = (const float*)d_in[1];
  const float* bqkv  = (const float*)d_in[2];
  const float* wattn = (const float*)d_in[3];
  const float* battn = (const float*)d_in[4];
  const float* wout  = (const float*)d_in[5];
  const float* bout  = (const float*)d_in[6];
  const float* relh  = (const float*)d_in[7];
  const float* relw  = (const float*)d_in[8];

  char* ws = (char*)d_ws;
  unsigned short* xT     = (unsigned short*)(ws + 0);         // 2 MB
  unsigned short* qbuf   = (unsigned short*)(ws + 2097152);   // 2 MB
  unsigned short* kbuf   = (unsigned short*)(ws + 4194304);   // 2 MB
  unsigned short* vbuf   = (unsigned short*)(ws + 6291456);   // 2 MB
  unsigned short* attnT  = (unsigned short*)(ws + 8388608);   // 2 MB
  unsigned short* wqkvb  = (unsigned short*)(ws + 10485760);  // 384 KB
  unsigned short* wattnb = (unsigned short*)(ws + 10878976);  // 128 KB
  unsigned short* woutb  = (unsigned short*)(ws + 11010048);  // 128 KB
  unsigned short* relhb  = (unsigned short*)(ws + 11141120);  // 4 KB
  unsigned short* relwb  = (unsigned short*)(ws + 11145216);  // 4 KB

  k_cvt_xT<<<4096, 256, 0, stream>>>(x, xT);
  k_cvt<<<768, 256, 0, stream>>>(wqkv, wqkvb, 768 * 256);
  k_cvt<<<256, 256, 0, stream>>>(wattn, wattnb, 256 * 256);
  k_cvt<<<256, 256, 0, stream>>>(wout, woutb, 256 * 256);
  k_cvt_rel<<<8, 256, 0, stream>>>(relh, relhb);
  k_cvt_rel<<<8, 256, 0, stream>>>(relw, relwb);

  k_qkv<<<dim3(12, 64), 128, 0, stream>>>(xT, wqkvb, bqkv, qbuf, kbuf, vbuf);
  k_attn<<<dim3(64, 8, 4), 32, 0, stream>>>(qbuf, kbuf, vbuf, relhb, relwb, attnT);
  k_out<<<dim3(8, 64), 128, 0, stream>>>(xT, attnT, woutb, wattnb, bout, battn,
                                         (float*)d_out);
}
